// IterativeRefinementModule_17085379904363
// MI455X (gfx1250) — matrix operand order repaired and checked by reading, not compiled, not run on hardware
//
#include <hip/hip_runtime.h>
#include <hip/hip_bf16.h>
#include <math.h>

// ---------------------------------------------------------------------------
// CDNA5 (gfx1250) iterative-refinement IK kernel.
// MLP (28->128->128->7, GELU+LayerNorm) on V_WMMA_F32_16X16X32_F16;
// FK / projection / LN reductions on wave32 VALU + __shfl_xor.
// ---------------------------------------------------------------------------

typedef __attribute__((ext_vector_type(16))) _Float16 v16h;
typedef __attribute__((ext_vector_type(8)))  _Float16 v8h;
typedef __attribute__((ext_vector_type(8)))  float    v8f;

union ABfrag { v16h v; v8h h2[2]; _Float16 e[16]; };

#define WAVES_PER_BLOCK 4
#define ROWS_PER_BLOCK  (WAVES_PER_BLOCK * 16)
#define NUM_ITER        3

__device__ __constant__ float LIM_LO[7] = {-2.8973f,-1.7628f,-2.8973f,-3.0718f,-2.8973f,-0.0175f,-2.8973f};
__device__ __constant__ float LIM_HI[7] = { 2.8973f, 1.7628f, 2.8973f,-0.0698f, 2.8973f, 3.7525f, 2.8973f};

// All _Float16 arrays first (keeps 32B alignment for v16h B-fragment loads).
struct __attribute__((aligned(32))) Smem {
  _Float16 W1T[128 * 32];                 // [n][k], k>=28 zero-padded     8 KB
  _Float16 W2T[128 * 128];                // [n][k]                       32 KB
  _Float16 W3T[16 * 128];                 // [n][k], n>=7 zero-padded      4 KB
  _Float16 H[WAVES_PER_BLOCK][16 * 128];  // per-wave activations         16 KB
  float b1[128], g1[128], be1[128];
  float b2[128], g2[128], be2[128];
  float b3v[16];
};                                        // total ~63.1 KB

__device__ __forceinline__ float gelu_exact(float x) {
  return 0.5f * x * (1.0f + erff(x * 0.70710678118654752f));
}

// Panda FK: 7 revolute-Z joints with fixed parent transforms (rx in {0,±pi/2}).
// Keypoints = translations of ts[0,2,3,4,6,7] and the j8/hand frame.
__device__ __forceinline__ void panda_fk(const float th[7], float kp[7][3]) {
  const float CXc[7] = {1.f, 0.f, 0.f, 0.f, 0.f, 0.f, 0.f};
  const float SXc[7] = {0.f,-1.f, 1.f, 1.f,-1.f, 1.f, 1.f};
  const float FTX[7] = {0.f, 0.f, 0.f, 0.0825f,-0.0825f, 0.f, 0.088f};
  const float FTY[7] = {0.f, 0.f,-0.316f, 0.f, 0.384f, 0.f, 0.f};
  const float FTZ[7] = {0.333f, 0.f, 0.f, 0.f, 0.f, 0.f, 0.f};
  float R00=1.f,R01=0.f,R02=0.f, R10=0.f,R11=1.f,R12=0.f, R20=0.f,R21=0.f,R22=1.f;
  float tx=0.f, ty=0.f, tz=0.f;
  kp[0][0]=0.f; kp[0][1]=0.f; kp[0][2]=0.f;
#pragma unroll
  for (int i = 0; i < 7; ++i) {
    float s, c; sincosf(th[i], &s, &c);
    const float cx = CXc[i], sx = SXc[i];
    // t' = R * F_t + t
    const float ntx = tx + R00*FTX[i] + R01*FTY[i] + R02*FTZ[i];
    const float nty = ty + R10*FTX[i] + R11*FTY[i] + R12*FTZ[i];
    const float ntz = tz + R20*FTX[i] + R21*FTY[i] + R22*FTZ[i];
    // step rotation S = Rx(fixed) * Rz(theta):
    // col0=(c, cx*s, sx*s)  col1=(-s, cx*c, sx*c)  col2=(0, -sx, cx)
    const float a0=c,  a1=cx*s, a2=sx*s;
    const float b0=-s, b1=cx*c, b2=sx*c;
    const float n00=R00*a0+R01*a1+R02*a2, n10=R10*a0+R11*a1+R12*a2, n20=R20*a0+R21*a1+R22*a2;
    const float n01=R00*b0+R01*b1+R02*b2, n11=R10*b0+R11*b1+R12*b2, n21=R20*b0+R21*b1+R22*b2;
    const float n02=-R01*sx+R02*cx,       n12=-R11*sx+R12*cx,       n22=-R21*sx+R22*cx;
    R00=n00; R01=n01; R02=n02; R10=n10; R11=n11; R12=n12; R20=n20; R21=n21; R22=n22;
    tx=ntx; ty=nty; tz=ntz;
    if (i==1){kp[1][0]=tx;kp[1][1]=ty;kp[1][2]=tz;}
    if (i==2){kp[2][0]=tx;kp[2][1]=ty;kp[2][2]=tz;}
    if (i==3){kp[3][0]=tx;kp[3][1]=ty;kp[3][2]=tz;}
    if (i==5){kp[4][0]=tx;kp[4][1]=ty;kp[4][2]=tz;}
    if (i==6){kp[5][0]=tx;kp[5][1]=ty;kp[5][2]=tz;}
  }
  // hand keypoint: T_J8 translates 0.107 along local z; T_HAND is pure Rz.
  kp[6][0] = tx + 0.107f * R02;
  kp[6][1] = ty + 0.107f * R12;
  kp[6][2] = tz + 0.107f * R22;
}

// Bias + exact GELU on C-fragments, LayerNorm over 128 features
// (8 N-tiles in-lane + shfl_xor butterfly across the 16-lane half),
// then store f16 activations row-major to the per-wave LDS H buffer.
__device__ __forceinline__ void bias_gelu_ln_store(
    v8f acc[8], const float* __restrict__ bb, const float* __restrict__ gg,
    const float* __restrict__ bet, _Float16* __restrict__ Hw, int lane) {
  const int lm = lane & 15, hh = lane >> 4;
  float vals[8][8];
  float s1[8], s2[8];
#pragma unroll
  for (int v = 0; v < 8; ++v) { s1[v] = 0.f; s2[v] = 0.f; }
#pragma unroll
  for (int nt = 0; nt < 8; ++nt) {
    const float bv = bb[nt * 16 + lm];
#pragma unroll
    for (int v = 0; v < 8; ++v) {
      const float y = gelu_exact(acc[nt][v] + bv);
      vals[nt][v] = y;
      s1[v] += y;
      s2[v] += y * y;
    }
  }
#pragma unroll
  for (int m = 1; m <= 8; m <<= 1) {  // stays within the 16-lane half
#pragma unroll
    for (int v = 0; v < 8; ++v) {
      s1[v] += __shfl_xor(s1[v], m, 32);
      s2[v] += __shfl_xor(s2[v], m, 32);
    }
  }
#pragma unroll
  for (int v = 0; v < 8; ++v) {
    const float mu  = s1[v] * (1.0f / 128.0f);
    const float var = s2[v] * (1.0f / 128.0f) - mu * mu;
    s1[v] = mu;
    s2[v] = rsqrtf(var + 1e-5f);
  }
#pragma unroll
  for (int nt = 0; nt < 8; ++nt) {
    const int f = nt * 16 + lm;
    const float gf = gg[f], bf = bet[f];
#pragma unroll
    for (int v = 0; v < 8; ++v) {
      const float y = (vals[nt][v] - s1[v]) * s2[v] * gf + bf;
      Hw[(8 * hh + v) * 128 + f] = (_Float16)y;  // C layout: row M = 8h+v
    }
  }
}

__global__ __launch_bounds__(WAVES_PER_BLOCK * 32)
void ik_refine_kernel(const float* __restrict__ th0,
                      const float* __restrict__ tgt2d,
                      const float* __restrict__ camK,
                      const float* __restrict__ Rext,
                      const float* __restrict__ text,
                      const unsigned char* __restrict__ vmask,
                      const float* __restrict__ W1, const float* __restrict__ b1,
                      const float* __restrict__ g1, const float* __restrict__ be1,
                      const float* __restrict__ W2, const float* __restrict__ b2,
                      const float* __restrict__ g2, const float* __restrict__ be2,
                      const float* __restrict__ W3, const float* __restrict__ b3,
                      const float* __restrict__ slog,
                      float* __restrict__ out, int nB) {
  __shared__ Smem sm;
  const int tid  = threadIdx.x;
  const int lane = tid & 31;
  const int wave = tid >> 5;
  const int lm   = lane & 15;
  const int hh   = lane >> 4;

  // Warm the weight stream (gfx1250 global_prefetch_b8).
  __builtin_prefetch(W2 + tid * 64, 0, 0);
  __builtin_prefetch(W1 + tid * 16, 0, 0);

  // ---- Stage weights to LDS as f16, column-major [n][k] so each lane's
  // B-fragment (K = 8*half..+7 and 16+8*half..+7 at fixed column n=lane&15) is two
  // contiguous 16B LDS loads. ----
  for (int i = tid; i < 128 * 32; i += blockDim.x) {
    const int n = i >> 5, k = i & 31;
    sm.W1T[i] = (k < 28) ? (_Float16)W1[k * 128 + n] : (_Float16)0.0f;
  }
  for (int i = tid; i < 128 * 128; i += blockDim.x) {
    const int n = i >> 7, k = i & 127;
    sm.W2T[i] = (_Float16)W2[k * 128 + n];
  }
  for (int i = tid; i < 16 * 128; i += blockDim.x) {
    const int n = i >> 7, k = i & 127;
    sm.W3T[i] = (n < 7) ? (_Float16)W3[k * 7 + n] : (_Float16)0.0f;
  }
  for (int i = tid; i < 128; i += blockDim.x) {
    sm.b1[i] = b1[i]; sm.g1[i] = g1[i]; sm.be1[i] = be1[i];
    sm.b2[i] = b2[i]; sm.g2[i] = g2[i]; sm.be2[i] = be2[i];
  }
  if (tid < 16) sm.b3v[tid] = (tid < 7) ? b3[tid] : 0.0f;
  __syncthreads();

  // ---- Per-row state. Both lane halves redundantly own row lm (this is
  // exactly the 16-bit A-fragment layout: halves share M, split K). ----
  const int  row    = blockIdx.x * ROWS_PER_BLOCK + wave * 16 + lm;
  const bool rvalid = row < nB;
  const int  rc     = rvalid ? row : 0;

  float Kc[9], Re[9], te[3], tg[14], theta[7];
#pragma unroll
  for (int i = 0; i < 9; ++i) {
    Kc[i] = camK[(size_t)rc * 9 + i];
    Re[i] = Rext[(size_t)rc * 9 + i];
  }
#pragma unroll
  for (int i = 0; i < 3; ++i) te[i] = text[(size_t)rc * 3 + i];
#pragma unroll
  for (int i = 0; i < 14; ++i) tg[i] = tgt2d[(size_t)rc * 14 + i];
#pragma unroll
  for (int j = 0; j < 7; ++j) theta[j] = th0[(size_t)rc * 7 + j];
  const float vm = vmask[rc] ? 1.0f : 0.0f;

  float kp[7][3];
  panda_fk(theta, kp);

  const size_t o1 = (size_t)7 * nB;   // kp_final
  const size_t o2 = (size_t)28 * nB;  // all_angles (4,B,7)
  const size_t o3 = (size_t)56 * nB;  // all_kp     (4,B,7,3)

  auto write_iter = [&](int it) {
    if (hh == 0 && rvalid) {
      const size_t ab = o2 + ((size_t)it * nB + row) * 7;
#pragma unroll
      for (int j = 0; j < 7; ++j) out[ab + j] = theta[j];
      const size_t kb = o3 + ((size_t)it * nB + row) * 21;
#pragma unroll
      for (int k = 0; k < 7; ++k)
#pragma unroll
        for (int c = 0; c < 3; ++c) out[kb + k * 3 + c] = kp[k][c];
    }
  };
  write_iter(0);

  _Float16* Hw   = sm.H[wave];
  float*    dbuf = reinterpret_cast<float*>(Hw);  // reused for delta transpose

  for (int it = 0; it < NUM_ITER; ++it) {
    // ---- Project keypoints, build 28-feature vector (pad K to 32). ----
    float x[32];
#pragma unroll
    for (int k = 0; k < 7; ++k) {
      const float cxx = Re[0]*kp[k][0] + Re[1]*kp[k][1] + Re[2]*kp[k][2] + te[0];
      const float cyy = Re[3]*kp[k][0] + Re[4]*kp[k][1] + Re[5]*kp[k][2] + te[1];
      const float czz = Re[6]*kp[k][0] + Re[7]*kp[k][1] + Re[8]*kp[k][2] + te[2];
      const float z   = fmaxf(czz, 1e-6f);
      const float inv = 1.0f / z;
      const float nx = cxx * inv, ny = cyy * inv, nz = czz * inv;
      const float u  = Kc[0]*nx + Kc[1]*ny + Kc[2]*nz;
      const float v  = Kc[3]*nx + Kc[4]*ny + Kc[5]*nz;
      const float d0 = (tg[2*k]   - u) * vm;
      const float d1 = (tg[2*k+1] - v) * vm;
      x[2*k]   = d0;
      x[2*k+1] = d1;
      x[14+k]  = theta[k];
      x[21+k]  = sqrtf(d0*d0 + d1*d1);
    }
    x[28] = x[29] = x[30] = x[31] = 0.f;

    // ---- Layer 1: X(16x32) @ W1(32x128) -> 8 WMMA tiles. A built in-reg
    // per the ISA A-layout: v<4 -> K=8h+2v, v>=4 -> K=16+8h+2(v-4). ----
    ABfrag a1;
#pragma unroll
    for (int v = 0; v < 8; ++v) {
      const int k0 = (v < 4) ? (8*hh + 2*v) : (16 + 8*hh + 2*(v-4));
      a1.e[2*v]   = (_Float16)x[k0];
      a1.e[2*v+1] = (_Float16)x[k0 + 1];
    }
    v8f acc[8];
#pragma unroll
    for (int nt = 0; nt < 8; ++nt) {
      ABfrag b;
      b.h2[0] = *reinterpret_cast<const v8h*>(&sm.W1T[(nt*16 + lm) * 32 + 8*hh]);
      b.h2[1] = *reinterpret_cast<const v8h*>(&sm.W1T[(nt*16 + lm) * 32 + 16 + 8*hh]);
      v8f z8 = {};
      acc[nt] = __builtin_amdgcn_wmma_f32_16x16x32_f16(
          false, a1.v, false, b.v, (short)0, z8, false, false);
    }
    bias_gelu_ln_store(acc, sm.b1, sm.g1, sm.be1, Hw, lane);
    __syncthreads();

    // ---- Layer 2: H(16x128) @ W2(128x128), K in 4 chunks of 32. ----
    ABfrag a2[4];
#pragma unroll
    for (int c = 0; c < 4; ++c) {
      a2[c].h2[0] = *reinterpret_cast<const v8h*>(&Hw[lm*128 + c*32 + 8*hh]);
      a2[c].h2[1] = *reinterpret_cast<const v8h*>(&Hw[lm*128 + c*32 + 16 + 8*hh]);
    }
    __syncthreads();  // all A reads done before Hw is overwritten below
#pragma unroll
    for (int nt = 0; nt < 8; ++nt) {
      v8f a = {};
#pragma unroll
      for (int c = 0; c < 4; ++c) {
        ABfrag b;
        b.h2[0] = *reinterpret_cast<const v8h*>(&sm.W2T[(nt*16 + lm) * 128 + c*32 + 8*hh]);
        b.h2[1] = *reinterpret_cast<const v8h*>(&sm.W2T[(nt*16 + lm) * 128 + c*32 + 16 + 8*hh]);
        a = __builtin_amdgcn_wmma_f32_16x16x32_f16(
            false, a2[c].v, false, b.v, (short)0, a, false, false);
      }
      acc[nt] = a;
    }
    bias_gelu_ln_store(acc, sm.b2, sm.g2, sm.be2, Hw, lane);
    __syncthreads();

    // ---- Layer 3: H(16x128) @ W3(128x16-padded), one N-tile. ----
    ABfrag a3[4];
#pragma unroll
    for (int c = 0; c < 4; ++c) {
      a3[c].h2[0] = *reinterpret_cast<const v8h*>(&Hw[lm*128 + c*32 + 8*hh]);
      a3[c].h2[1] = *reinterpret_cast<const v8h*>(&Hw[lm*128 + c*32 + 16 + 8*hh]);
    }
    __syncthreads();
    v8f d8 = {};
#pragma unroll
    for (int c = 0; c < 4; ++c) {
      ABfrag b;
      b.h2[0] = *reinterpret_cast<const v8h*>(&sm.W3T[lm * 128 + c*32 + 8*hh]);
      b.h2[1] = *reinterpret_cast<const v8h*>(&sm.W3T[lm * 128 + c*32 + 16 + 8*hh]);
      d8 = __builtin_amdgcn_wmma_f32_16x16x32_f16(
          false, a3[c].v, false, b.v, (short)0, d8, false, false);
    }
    // Transpose delta (C-fragment) -> row-per-lane through LDS.
#pragma unroll
    for (int v = 0; v < 8; ++v)
      dbuf[(8*hh + v) * 16 + lm] = d8[v] + sm.b3v[lm];
    __syncthreads();

    const float ss = 1.0f / (1.0f + expf(-slog[it]));
#pragma unroll
    for (int j = 0; j < 7; ++j) {
      const float nt = theta[j] + ss * dbuf[lm * 16 + j];
      theta[j] = fminf(fmaxf(nt, LIM_LO[j]), LIM_HI[j]);
    }
    __syncthreads();  // dbuf reads complete before next iteration reuses Hw

    panda_fk(theta, kp);
    write_iter(it + 1);
  }

  // ---- Final theta + keypoints. ----
  if (hh == 0 && rvalid) {
#pragma unroll
    for (int j = 0; j < 7; ++j) out[(size_t)row * 7 + j] = theta[j];
#pragma unroll
    for (int k = 0; k < 7; ++k)
#pragma unroll
      for (int c = 0; c < 3; ++c)
        out[o1 + (size_t)row * 21 + k * 3 + c] = kp[k][c];
  }
}

extern "C" void kernel_launch(void* const* d_in, const int* in_sizes, int n_in,
                              void* d_out, int out_size, void* d_ws, size_t ws_size,
                              hipStream_t stream) {
  (void)n_in; (void)out_size; (void)d_ws; (void)ws_size;
  const float*         th0  = (const float*)d_in[0];
  const float*         tgt  = (const float*)d_in[1];
  const float*         camK = (const float*)d_in[2];
  // d_in[3] = original_size (unused by the reference forward)
  const float*         Rext = (const float*)d_in[4];
  const float*         text = (const float*)d_in[5];
  const unsigned char* vm   = (const unsigned char*)d_in[6];
  const float* W1  = (const float*)d_in[7];
  const float* b1  = (const float*)d_in[8];
  const float* g1  = (const float*)d_in[9];
  const float* be1 = (const float*)d_in[10];
  const float* W2  = (const float*)d_in[11];
  const float* b2  = (const float*)d_in[12];
  const float* g2  = (const float*)d_in[13];
  const float* be2 = (const float*)d_in[14];
  const float* W3  = (const float*)d_in[15];
  const float* b3  = (const float*)d_in[16];
  const float* sl  = (const float*)d_in[17];

  const int nB = in_sizes[0] / 7;
  dim3 grid((nB + ROWS_PER_BLOCK - 1) / ROWS_PER_BLOCK);
  dim3 block(WAVES_PER_BLOCK * 32);
  ik_refine_kernel<<<grid, block, 0, stream>>>(
      th0, tgt, camK, Rext, text, vm,
      W1, b1, g1, be1, W2, b2, g2, be2, W3, b3, sl,
      (float*)d_out, nB);
}
